// FlowLSTM_3788161155276
// MI455X (gfx1250) — matrix operand order repaired and checked by reading, not compiled, not run on hardware
//
#include <hip/hip_runtime.h>

#define HID    1024
#define NGATE  4096     // 4 * HID
#define BATCH  4096
#define TSTEPS 19
#define INP    17
#define MT     16       // batch rows per workgroup
#define HROW   1032     // padded LDS row stride (bf16 elems): bank spread, rows stay 16B aligned
#define XROW   40       // padded x-tile row stride
#define NTHREADS 1024   // 32 waves (max WG size; full wave-slot occupancy on a WGP)
#define NWAVES  (NTHREADS / 32)
#define CPW     (64 / NWAVES)   // 16-col hidden chunks per wave = 2

typedef __attribute__((ext_vector_type(16))) __bf16 v16bf;
typedef __attribute__((ext_vector_type(8)))  __bf16 v8bf;
typedef __attribute__((ext_vector_type(8)))  float  v8f;

__device__ __forceinline__ float sigm(float x) {
    return 1.0f / (1.0f + __expf(-x));
}
__device__ __forceinline__ float tanh_fast(float x) {
    x = fminf(fmaxf(x, -15.0f), 15.0f);
    float e = __expf(-2.0f * x);
    return (1.0f - e) / (1.0f + e);
}

// A fragment pieces at (per-lane base + elem offset). 16B contiguous loads.
// ISA A layout (16x32 bf16): lanes 0-15 (M=lane): K=k0+0..7, k0+16..23
//                            lanes 16-31 (M=lane-16): K=k0+8..15, k0+24..31
__device__ __forceinline__ v16bf load_a_at(const __bf16* p, int off) {
    union { v16bf v; v8bf h[2]; } u;
    u.h[0] = *(const v8bf*)(p + off);
    u.h[1] = *(const v8bf*)(p + off + 16);
    return u.v;
}
// B fragment pieces: ISA B layout (32x16): lanes 0-15 (N=lane): K=k0+0..7, k0+16..23,
// lanes 16-31: K=k0+8..15, k0+24..31 -> two 16B loads along W row.
__device__ __forceinline__ v16bf load_b_at(const __bf16* __restrict__ p, int off) {
    union { v16bf v; v8bf h[2]; } u;
    u.h[0] = *(const v8bf*)(p + off);
    u.h[1] = *(const v8bf*)(p + off + 16);
    return u.v;
}

__device__ __forceinline__ v8f wmma_bf16(v16bf a, v16bf b, v8f c) {
    return __builtin_amdgcn_wmma_f32_16x16x32_bf16(
        /*neg_a=*/false, a, /*neg_b=*/false, b,
        /*c_mod=*/(short)0, c, /*reuse_a=*/false, /*reuse_b=*/false);
}

// K loop, 2x unrolled with ping-pong fragment sets reloaded IN PLACE
// (no register rotation copies). Next half's 8 global + 2 LDS loads are in
// flight while the current half's 4 WMMAs issue.
template <int KTOT>
__device__ __forceinline__ void gemm_k(const __bf16* aBase, int aStride,
                                       const __bf16* __restrict__ W, int rowlen,
                                       int lane, int nbase, v8f acc[4]) {
    const __bf16* ap = aBase + (lane & 15) * aStride + ((lane >> 4) << 3);
    const __bf16* __restrict__ wp =
        W + (size_t)(nbase + (lane & 15)) * rowlen + ((lane >> 4) << 3);
    const int gs = HID * rowlen;   // gate-block stride (compile-time const)

    v16bf a0  = load_a_at(ap, 0);
    v16bf b00 = load_b_at(wp, 0 * gs);
    v16bf b01 = load_b_at(wp, 1 * gs);
    v16bf b02 = load_b_at(wp, 2 * gs);
    v16bf b03 = load_b_at(wp, 3 * gs);

    int k0 = 32;
    #pragma unroll 1
    for (; k0 + 32 < KTOT; k0 += 64) {
        v16bf a1  = load_a_at(ap, k0);
        v16bf b10 = load_b_at(wp, 0 * gs + k0);
        v16bf b11 = load_b_at(wp, 1 * gs + k0);
        v16bf b12 = load_b_at(wp, 2 * gs + k0);
        v16bf b13 = load_b_at(wp, 3 * gs + k0);
        acc[0] = wmma_bf16(a0, b00, acc[0]);
        acc[1] = wmma_bf16(a0, b01, acc[1]);
        acc[2] = wmma_bf16(a0, b02, acc[2]);
        acc[3] = wmma_bf16(a0, b03, acc[3]);
        a0  = load_a_at(ap, k0 + 32);
        b00 = load_b_at(wp, 0 * gs + k0 + 32);
        b01 = load_b_at(wp, 1 * gs + k0 + 32);
        b02 = load_b_at(wp, 2 * gs + k0 + 32);
        b03 = load_b_at(wp, 3 * gs + k0 + 32);
        acc[0] = wmma_bf16(a1, b10, acc[0]);
        acc[1] = wmma_bf16(a1, b11, acc[1]);
        acc[2] = wmma_bf16(a1, b12, acc[2]);
        acc[3] = wmma_bf16(a1, b13, acc[3]);
    }
    if (k0 < KTOT) {               // one remaining K-step beyond the preloaded one
        v16bf a1  = load_a_at(ap, k0);
        v16bf b10 = load_b_at(wp, 0 * gs + k0);
        v16bf b11 = load_b_at(wp, 1 * gs + k0);
        v16bf b12 = load_b_at(wp, 2 * gs + k0);
        v16bf b13 = load_b_at(wp, 3 * gs + k0);
        acc[0] = wmma_bf16(a0, b00, acc[0]);
        acc[1] = wmma_bf16(a0, b01, acc[1]);
        acc[2] = wmma_bf16(a0, b02, acc[2]);
        acc[3] = wmma_bf16(a0, b03, acc[3]);
        acc[0] = wmma_bf16(a1, b10, acc[0]);
        acc[1] = wmma_bf16(a1, b11, acc[1]);
        acc[2] = wmma_bf16(a1, b12, acc[2]);
        acc[3] = wmma_bf16(a1, b13, acc[3]);
    } else {
        acc[0] = wmma_bf16(a0, b00, acc[0]);
        acc[1] = wmma_bf16(a0, b01, acc[1]);
        acc[2] = wmma_bf16(a0, b02, acc[2]);
        acc[3] = wmma_bf16(a0, b03, acc[3]);
    }
}

// Elementwise LSTM cell on WMMA accumulator layout; c stays in VGPRs, h -> LDS bf16.
// D layout: vgpr v, lanes 0-15: (M=v, N=lane); lanes 16-31: (M=v+8, N=lane-16)
__device__ __forceinline__ void lstm_act_store(v8f acc[4], float cs[8], __bf16* hout,
                                               int lane, int n0) {
    int col   = n0 + (lane & 15);
    int mbase = (lane >> 4) << 3;
    #pragma unroll
    for (int v = 0; v < 8; ++v) {
        float ig = sigm(acc[0][v]);
        float fg = sigm(acc[1][v]);
        float gg = tanh_fast(acc[2][v]);
        float og = sigm(acc[3][v]);
        float c  = fg * cs[v] + ig * gg;
        cs[v]    = c;
        float h  = og * tanh_fast(c);
        hout[(mbase + v) * HROW + col] = (__bf16)h;
    }
}

// ---------------- prep: f32 weights -> bf16 (and pad W_ih1 to K=32), fuse biases ----
__global__ __launch_bounds__(256)
void lstm_prep_kernel(const float* __restrict__ Wih1f, const float* __restrict__ Whh1f,
                      const float* __restrict__ bih1,  const float* __restrict__ bhh1,
                      const float* __restrict__ Wih2f, const float* __restrict__ Whh2f,
                      const float* __restrict__ bih2,  const float* __restrict__ bhh2,
                      __bf16* __restrict__ Whh1, __bf16* __restrict__ Wih2,
                      __bf16* __restrict__ Whh2, __bf16* __restrict__ Wih1p,
                      float* __restrict__ bias1, float* __restrict__ bias2) {
    int j = blockIdx.x;             // gate row 0..4095
    int t = threadIdx.x;
    for (int k = t; k < HID; k += 256) {
        size_t o = (size_t)j * HID + k;
        Whh1[o] = (__bf16)Whh1f[o];
        Wih2[o] = (__bf16)Wih2f[o];
        Whh2[o] = (__bf16)Whh2f[o];
    }
    if (t < 32)
        Wih1p[j * 32 + t] = (t < INP) ? (__bf16)Wih1f[j * INP + t] : (__bf16)0.0f;
    if (t == 0) {
        bias1[j] = bih1[j] + bhh1[j];
        bias2[j] = bih2[j] + bhh2[j];
    }
}

// ---------------- main: persistent per-batch-tile kernel over all T steps ----------
__global__ __launch_bounds__(NTHREADS, 1)
void lstm_seq_kernel(const float* __restrict__ x,
                     const __bf16* __restrict__ Whh1,
                     const __bf16* __restrict__ Wih2,
                     const __bf16* __restrict__ Whh2,
                     const __bf16* __restrict__ Wih1p,
                     const float* __restrict__ bias1,
                     const float* __restrict__ bias2,
                     const float* __restrict__ Wlin,
                     const float* __restrict__ blin,
                     float* __restrict__ out) {
    __shared__ __bf16 sH1[2][MT * HROW];   // double-buffered h1 (bf16)
    __shared__ __bf16 sH2[2][MT * HROW];   // double-buffered h2 (bf16)
    __shared__ __bf16 sX[MT * XROW];       // x_t tile padded to K=32

    const int tid  = threadIdx.x;
    const int lane = tid & 31;
    const int wid  = tid >> 5;
    const int row0 = blockIdx.x * MT;

    for (int i = tid; i < MT * HROW; i += NTHREADS) {
        sH1[0][i] = (__bf16)0.0f; sH1[1][i] = (__bf16)0.0f;
        sH2[0][i] = (__bf16)0.0f; sH2[1][i] = (__bf16)0.0f;
    }
    float c1s[CPW][8];
    float c2s[CPW][8];
    #pragma unroll
    for (int j = 0; j < CPW; ++j)
        #pragma unroll
        for (int v = 0; v < 8; ++v) { c1s[j][v] = 0.0f; c2s[j][v] = 0.0f; }
    __syncthreads();

    for (int t = 0; t < TSTEPS; ++t) {
        const int p = t & 1;
        const __bf16* h1old = sH1[p];
        __bf16*       h1new = sH1[p ^ 1];
        const __bf16* h2old = sH2[p];
        __bf16*       h2new = sH2[p ^ 1];

        // stage x_t (16 x 17 -> 16 x 32, zero padded), f32 -> bf16
        for (int idx = tid; idx < MT * 32; idx += NTHREADS) {
            int r = idx >> 5, c = idx & 31;
            float xv = (c < INP) ? x[((size_t)(row0 + r) * TSTEPS + t) * INP + c] : 0.0f;
            sX[r * XROW + c] = (__bf16)xv;
        }
        __syncthreads();

        // ---------------- layer 1: gates = h1_old @ Whh1^T + x_t @ Wih1^T + b ----
        #pragma unroll
        for (int j = 0; j < CPW; ++j) {
            const int n0 = (wid * CPW + j) * 16;
            v8f acc[4];
            #pragma unroll
            for (int g = 0; g < 4; ++g) {
                float bv = bias1[g * HID + n0 + (lane & 15)];
                #pragma unroll
                for (int v = 0; v < 8; ++v) acc[g][v] = bv;
            }
            gemm_k<HID>(h1old, HROW, Whh1, HID, lane, n0, acc);
            gemm_k<32>(sX, XROW, Wih1p, 32, lane, n0, acc);
            lstm_act_store(acc, c1s[j], h1new, lane, n0);
        }
        __syncthreads();

        // ---------------- layer 2: gates = h1_new @ Wih2^T + h2_old @ Whh2^T + b --
        #pragma unroll
        for (int j = 0; j < CPW; ++j) {
            const int n0 = (wid * CPW + j) * 16;
            v8f acc[4];
            #pragma unroll
            for (int g = 0; g < 4; ++g) {
                float bv = bias2[g * HID + n0 + (lane & 15)];
                #pragma unroll
                for (int v = 0; v < 8; ++v) acc[g][v] = bv;
            }
            gemm_k<HID>(h1new, HROW, Wih2, HID, lane, n0, acc);
            gemm_k<HID>(h2old, HROW, Whh2, HID, lane, n0, acc);
            lstm_act_store(acc, c2s[j], h2new, lane, n0);
        }
        __syncthreads();

        // ---------------- head: out[b,t,:] = h2_new @ Wlin^T + blin (17 wide) -----
        for (int idx = tid; idx < MT * INP; idx += NTHREADS) {
            int r = idx / INP, c = idx - r * INP;
            const __bf16* hp = h2new + r * HROW;
            const float*  wl = Wlin + c * HID;
            float s = blin[c];
            #pragma unroll 8
            for (int k = 0; k < HID; ++k) s += (float)hp[k] * wl[k];
            out[((size_t)(row0 + r) * TSTEPS + t) * INP + c] = s;
        }
        __syncthreads();
    }
}

extern "C" void kernel_launch(void* const* d_in, const int* in_sizes, int n_in,
                              void* d_out, int out_size, void* d_ws, size_t ws_size,
                              hipStream_t stream) {
    const float* x     = (const float*)d_in[0];
    const float* Wih1f = (const float*)d_in[1];
    const float* Whh1f = (const float*)d_in[2];
    const float* bih1  = (const float*)d_in[3];
    const float* bhh1  = (const float*)d_in[4];
    const float* Wih2f = (const float*)d_in[5];
    const float* Whh2f = (const float*)d_in[6];
    const float* bih2  = (const float*)d_in[7];
    const float* bhh2  = (const float*)d_in[8];
    const float* Wlin  = (const float*)d_in[9];
    const float* blin  = (const float*)d_in[10];
    float* out = (float*)d_out;

    // workspace layout (bf16 weights resident in L2 across all timesteps): ~24.3 MB
    char* ws = (char*)d_ws;
    __bf16* Whh1  = (__bf16*)(ws);
    __bf16* Wih2  = (__bf16*)(ws + (size_t)NGATE * HID * 2);
    __bf16* Whh2  = (__bf16*)(ws + (size_t)NGATE * HID * 4);
    __bf16* Wih1p = (__bf16*)(ws + (size_t)NGATE * HID * 6);
    float*  bias1 = (float*) (ws + (size_t)NGATE * HID * 6 + (size_t)NGATE * 32 * 2);
    float*  bias2 = bias1 + NGATE;

    lstm_prep_kernel<<<NGATE, 256, 0, stream>>>(
        Wih1f, Whh1f, bih1, bhh1, Wih2f, Whh2f, bih2, bhh2,
        Whh1, Wih2, Whh2, Wih1p, bias1, bias2);

    lstm_seq_kernel<<<BATCH / MT, NTHREADS, 0, stream>>>(
        x, Whh1, Wih2, Whh2, Wih1p, bias1, bias2, Wlin, blin, out);
}
